// CustomRNN_72224170049941
// MI455X (gfx1250) — matrix operand order repaired and checked by reading, not compiled, not run on hardware
//
#include <hip/hip_runtime.h>
#include <hip/hip_bf16.h>
#include <stdint.h>
#include <stddef.h>

typedef __bf16 bf16;
typedef __attribute__((ext_vector_type(16))) __bf16 v16bf;
typedef __attribute__((ext_vector_type(8)))  __bf16 v8bf;
typedef __attribute__((ext_vector_type(8)))  float  v8f;

#define NB  32     // batch
#define TT  8192   // time steps
#define DXD 256    // state dim
#define DYD 128    // y dim
#define DUD 64     // u dim
#define DYH 128    // y_hat dim
#define DZH 64     // z_hat dim
#define CH  64     // chunk length S
#define NC  (TT/CH) // 128 chunks

#define WMMA_BF16(Af,Bf,Cf) \
  __builtin_amdgcn_wmma_f32_16x16x32_bf16(false,(Af),false,(Bf),(short)0,(Cf),false,false)

__device__ __forceinline__ v8f vzero8() {
  v8f z = {0.f,0.f,0.f,0.f,0.f,0.f,0.f,0.f};
  return z;
}

// A-operand fragment (16x32 bf16, M across lanes). Per-lane row pointer.
// ISA layout: lanes 0-15 hold K {0..7,16..23}; lanes 16-31 hold K {8..15,24..31}.
__device__ __forceinline__ v16bf fragA_bf(const bf16* row, int kt, int hi) {
  const int kb = kt*32 + hi*8;
  v8bf lo = *(const v8bf*)(row + kb);
  v8bf up = *(const v8bf*)(row + kb + 16);
  v16bf r;
  #pragma unroll
  for (int i = 0; i < 8; i++) { r[i] = lo[i]; r[i+8] = up[i]; }
  return r;
}

// Same A layout, but source is fp32 in global memory -> convert to bf16 in flight.
__device__ __forceinline__ v16bf fragA_f32(const float* row, int kloc, int hi) {
  const int kb = kloc + hi*8;
  v16bf r;
  #pragma unroll
  for (int i = 0; i < 8; i++) { r[i] = (bf16)row[kb+i]; r[i+8] = (bf16)row[kb+16+i]; }
  return r;
}

// B-operand fragment (32x16 bf16, N across lanes). rowBT points at row n of the
// transposed weight (BT layout, k contiguous). ISA layout: lanes 0-15 hold
// K {0..7,16..23}, lanes 16-31 hold K {8..15,24..31} of the kt window.
__device__ __forceinline__ v16bf fragB_bf(const bf16* rowBT, int kt, int hi) {
  const int kb = kt*32 + hi*8;
  v8bf lo = *(const v8bf*)(rowBT + kb);
  v8bf up = *(const v8bf*)(rowBT + kb + 16);
  v16bf r;
  #pragma unroll
  for (int i = 0; i < 8; i++) { r[i] = lo[i]; r[i+8] = up[i]; }
  return r;
}

// ---------------------------------------------------------------------------
// Weight staging: WXF[n][k] (n<256: k<256 = W_A, 256..383 = W_K, 384..447 = W_B)
// WYZ[j][k] (j<128 = W_Cy, j>=128 = W_Cz). biasX = b_A+b_K+b_B, biasYZ = b_Cy|b_Cz.
// Also seeds P0 = W_A (fp32) for the matrix-power squarings.
// ---------------------------------------------------------------------------
__global__ __launch_bounds__(448) void build_weights(
    const float* __restrict__ W_A, const float* __restrict__ b_A,
    const float* __restrict__ W_K, const float* __restrict__ b_K,
    const float* __restrict__ W_B, const float* __restrict__ b_B,
    const float* __restrict__ W_Cy, const float* __restrict__ b_Cy,
    const float* __restrict__ W_Cz, const float* __restrict__ b_Cz,
    bf16* __restrict__ WXF, bf16* __restrict__ WYZ,
    float* __restrict__ biasX, float* __restrict__ biasYZ,
    float* __restrict__ P0)
{
  const int n = blockIdx.x;
  const int k = threadIdx.x;
  if (n < DXD) {
    float v;
    if (k < DXD)        v = W_A[n*DXD + k];
    else if (k < DXD+DYD) v = W_K[n*DYD + (k - DXD)];
    else                v = W_B[n*DUD + (k - DXD - DYD)];
    WXF[n*448 + k] = (bf16)v;
    if (k < DXD) P0[n*DXD + k] = W_A[n*DXD + k];
    if (k == 0)  biasX[n] = b_A[n] + b_K[n] + b_B[n];
  } else {
    const int j = n - DXD; // 0..191
    if (k < DXD)
      WYZ[j*DXD + k] = (bf16)((j < DYH) ? W_Cy[j*DXD + k] : W_Cz[(j-DYH)*DXD + k]);
    if (k == 0)
      biasYZ[j] = (j < DYH) ? b_Cy[j] : b_Cz[j - DYH];
  }
}

// C = X * X (row-major, 256x256, fp32). Tiny (33 MFLOP) -> plain VALU.
__global__ __launch_bounds__(256) void square256(const float* __restrict__ X,
                                                 float* __restrict__ C)
{
  __shared__ float row[DXD];
  const int i = blockIdx.x, l = threadIdx.x;
  row[l] = X[i*DXD + l];
  __syncthreads();
  float acc = 0.f;
  #pragma unroll 8
  for (int j = 0; j < DXD; j++) acc = fmaf(row[j], X[j*DXD + l], acc);
  C[i*DXD + l] = acc;
}

__global__ __launch_bounds__(256) void cvt256(const float* __restrict__ P,
                                              bf16* __restrict__ M)
{
  const int idx = blockIdx.x*256 + threadIdx.x;
  M[idx] = (bf16)P[idx];
}

// ---------------------------------------------------------------------------
// Phase 1: per chunk c, run S=64 steps from x=0 with the fused step GEMM
//   x_next[32,256] = [x | y_t | u_t] (32x448) @ WXF^T  (+ biasX per step)
// and store the final state as D_c (fp32).
// 16 waves: wave w owns state n-tile w. Inner GEMM is all v_wmma bf16.
// ---------------------------------------------------------------------------
__global__ __launch_bounds__(512) void phase1(
    const float* __restrict__ y, const float* __restrict__ u,
    const bf16* __restrict__ WXF, const float* __restrict__ biasX,
    float* __restrict__ Dc)
{
  __shared__ bf16 sW[DXD*DXD]; // 131072 B: state-transition weights (BT layout)
  __shared__ bf16 sX[NB*DXD];  // 16384 B: state (bf16)
  const int tid = threadIdx.x;
  const int w = tid >> 5, lane = tid & 31, hi = lane >> 4, l15 = lane & 15;
  const int c = blockIdx.x;

  // stage sW: first 256 k of each WXF row (rows are 448 bf16 = 896 B, 16B-aligned)
  for (int i = tid; i < DXD*32; i += 512) {
    const int r = i >> 5, cc = i & 31;
    ((uint4*)sW)[i] = ((const uint4*)(WXF + (size_t)r*448))[cc];
  }
  for (int i = tid; i < NB*DXD/8; i += 512) ((uint4*)sX)[i] = make_uint4(0,0,0,0);

  // resident input-drive weight fragments (k-steps 8..13 of this wave's n-tile)
  const bf16* wrowG = WXF + (size_t)(w*16 + l15)*448;
  v16bf win[6];
  #pragma unroll
  for (int q = 0; q < 6; q++) win[q] = fragB_bf(wrowG, 8 + q, hi);
  const float bxv = biasX[w*16 + l15];
  __syncthreads();

  const bf16* xr0 = sX + l15*DXD;
  const bf16* xr1 = sX + (16 + l15)*DXD;
  const bf16* wr  = sW + (w*16 + l15)*DXD;

  for (int s = 0; s < CH; s++) {
    const int t = c*CH + s;
    const float* yr0 = y + ((size_t)l15*TT + t)*DYD;
    const float* yr1 = y + ((size_t)(l15+16)*TT + t)*DYD;
    const float* ur0 = u + ((size_t)l15*TT + t)*DUD;
    const float* ur1 = u + ((size_t)(l15+16)*TT + t)*DUD;

    v8f a0 = vzero8(), a1 = vzero8();
    #pragma unroll
    for (int kt = 0; kt < 8; kt++) {           // state part (K = 256)
      v16bf A0 = fragA_bf(xr0, kt, hi);
      v16bf A1 = fragA_bf(xr1, kt, hi);
      v16bf Bf = fragB_bf(wr, kt, hi);
      a0 = WMMA_BF16(A0, Bf, a0);
      a1 = WMMA_BF16(A1, Bf, a1);
    }
    #pragma unroll
    for (int kt = 0; kt < 4; kt++) {           // y drive (K = 128)
      v16bf A0 = fragA_f32(yr0, kt*32, hi);
      v16bf A1 = fragA_f32(yr1, kt*32, hi);
      a0 = WMMA_BF16(A0, win[kt], a0);
      a1 = WMMA_BF16(A1, win[kt], a1);
    }
    #pragma unroll
    for (int kt = 0; kt < 2; kt++) {           // u drive (K = 64)
      v16bf A0 = fragA_f32(ur0, kt*32, hi);
      v16bf A1 = fragA_f32(ur1, kt*32, hi);
      a0 = WMMA_BF16(A0, win[4+kt], a0);
      a1 = WMMA_BF16(A1, win[4+kt], a1);
    }
    __syncthreads(); // all state reads done before overwrite
    if (s == CH-1) {
      float* base = Dc + (size_t)c*NB*DXD;
      #pragma unroll
      for (int r = 0; r < 8; r++) {
        const int m = r + hi*8, n = w*16 + l15;
        base[(size_t)m*DXD + n]      = a0[r] + bxv;
        base[(size_t)(m+16)*DXD + n] = a1[r] + bxv;
      }
    } else {
      #pragma unroll
      for (int r = 0; r < 8; r++) {
        const int m = r + hi*8, n = w*16 + l15;
        sX[m*DXD + n]      = (bf16)(a0[r] + bxv);
        sX[(m+16)*DXD + n] = (bf16)(a1[r] + bxv);
      }
      __syncthreads();
    }
  }
}

// ---------------------------------------------------------------------------
// Combine: x_start[c+1] = x_start[c] @ (W_A^T)^S + D_c. MPT is (W_A)^S in
// row-major, which is exactly the BT layout this GEMM needs. Single WG.
// ---------------------------------------------------------------------------
__global__ __launch_bounds__(512) void combine(
    const bf16* __restrict__ MPT, const float* __restrict__ Dc,
    bf16* __restrict__ XS)
{
  __shared__ bf16 sW[DXD*DXD];
  __shared__ bf16 sX[NB*DXD];
  const int tid = threadIdx.x;
  const int w = tid >> 5, lane = tid & 31, hi = lane >> 4, l15 = lane & 15;

  for (int i = tid; i < DXD*32; i += 512) ((uint4*)sW)[i] = ((const uint4*)MPT)[i];
  for (int i = tid; i < NB*DXD/8; i += 512) ((uint4*)sX)[i] = make_uint4(0,0,0,0);
  __syncthreads();

  const bf16* xr0 = sX + l15*DXD;
  const bf16* xr1 = sX + (16 + l15)*DXD;
  const bf16* wr  = sW + (w*16 + l15)*DXD;

  for (int c = 0; c < NC; c++) {
    // snapshot current start state for phase 3
    bf16* xsd = XS + (size_t)c*NB*DXD;
    for (int i = tid; i < NB*DXD/8; i += 512) ((uint4*)xsd)[i] = ((const uint4*)sX)[i];

    v8f a0 = vzero8(), a1 = vzero8();
    #pragma unroll
    for (int kt = 0; kt < 8; kt++) {
      v16bf A0 = fragA_bf(xr0, kt, hi);
      v16bf A1 = fragA_bf(xr1, kt, hi);
      v16bf Bf = fragB_bf(wr, kt, hi);
      a0 = WMMA_BF16(A0, Bf, a0);
      a1 = WMMA_BF16(A1, Bf, a1);
    }
    __syncthreads();
    const float* dcb = Dc + (size_t)c*NB*DXD;
    #pragma unroll
    for (int r = 0; r < 8; r++) {
      const int m = r + hi*8, n = w*16 + l15;
      sX[m*DXD + n]      = (bf16)(a0[r] + dcb[(size_t)m*DXD + n]);
      sX[(m+16)*DXD + n] = (bf16)(a1[r] + dcb[(size_t)(m+16)*DXD + n]);
    }
    __syncthreads();
  }
}

// ---------------------------------------------------------------------------
// Phase 3: per chunk, re-run from corrected x_start[c], emitting y/z outputs
// from the pre-update state each step. Waves 0..15 own state n-tiles; waves
// 0..11 additionally own one 16-wide y/z projection tile (weights in VGPRs).
// ---------------------------------------------------------------------------
__global__ __launch_bounds__(512) void phase3(
    const float* __restrict__ y, const float* __restrict__ u,
    const bf16* __restrict__ WXF, const bf16* __restrict__ WYZ,
    const float* __restrict__ biasX, const float* __restrict__ biasYZ,
    const bf16* __restrict__ XS, float* __restrict__ outy)
{
  __shared__ bf16 sW[DXD*DXD];
  __shared__ bf16 sX[NB*DXD];
  const int tid = threadIdx.x;
  const int w = tid >> 5, lane = tid & 31, hi = lane >> 4, l15 = lane & 15;
  const int c = blockIdx.x;
  float* outz = outy + (size_t)NB*TT*DYH;

  for (int i = tid; i < DXD*32; i += 512) {
    const int r = i >> 5, cc = i & 31;
    ((uint4*)sW)[i] = ((const uint4*)(WXF + (size_t)r*448))[cc];
  }
  {
    const bf16* xs = XS + (size_t)c*NB*DXD;
    for (int i = tid; i < NB*DXD/8; i += 512) ((uint4*)sX)[i] = ((const uint4*)xs)[i];
  }

  const bf16* wrowG = WXF + (size_t)(w*16 + l15)*448;
  v16bf win[6];
  #pragma unroll
  for (int q = 0; q < 6; q++) win[q] = fragB_bf(wrowG, 8 + q, hi);

  const int wy = (w < 12) ? w : 0; // waves >=12 load-but-ignore (stay in bounds)
  const bf16* wzrow = WYZ + (size_t)(wy*16 + l15)*DXD;
  v16bf wyz[8];
  #pragma unroll
  for (int q = 0; q < 8; q++) wyz[q] = fragB_bf(wzrow, q, hi);

  const float bxv  = biasX[w*16 + l15];
  const float byzv = (w < 12) ? biasYZ[w*16 + l15] : 0.f;
  __syncthreads();

  const bf16* xr0 = sX + l15*DXD;
  const bf16* xr1 = sX + (16 + l15)*DXD;
  const bf16* wr  = sW + (w*16 + l15)*DXD;

  for (int s = 0; s < CH; s++) {
    const int t = c*CH + s;
    const float* yr0 = y + ((size_t)l15*TT + t)*DYD;
    const float* yr1 = y + ((size_t)(l15+16)*TT + t)*DYD;
    const float* ur0 = u + ((size_t)l15*TT + t)*DUD;
    const float* ur1 = u + ((size_t)(l15+16)*TT + t)*DUD;

    v8f aX0 = vzero8(), aX1 = vzero8(), aY0 = vzero8(), aY1 = vzero8();
    #pragma unroll
    for (int kt = 0; kt < 8; kt++) {
      v16bf A0 = fragA_bf(xr0, kt, hi);
      v16bf A1 = fragA_bf(xr1, kt, hi);
      v16bf Bx = fragB_bf(wr, kt, hi);
      aX0 = WMMA_BF16(A0, Bx, aX0);
      aX1 = WMMA_BF16(A1, Bx, aX1);
      if (w < 12) { // wave-uniform: EXEC stays all-ones
        aY0 = WMMA_BF16(A0, wyz[kt], aY0);
        aY1 = WMMA_BF16(A1, wyz[kt], aY1);
      }
    }
    #pragma unroll
    for (int kt = 0; kt < 4; kt++) {
      v16bf A0 = fragA_f32(yr0, kt*32, hi);
      v16bf A1 = fragA_f32(yr1, kt*32, hi);
      aX0 = WMMA_BF16(A0, win[kt], aX0);
      aX1 = WMMA_BF16(A1, win[kt], aX1);
    }
    #pragma unroll
    for (int kt = 0; kt < 2; kt++) {
      v16bf A0 = fragA_f32(ur0, kt*32, hi);
      v16bf A1 = fragA_f32(ur1, kt*32, hi);
      aX0 = WMMA_BF16(A0, win[4+kt], aX0);
      aX1 = WMMA_BF16(A1, win[4+kt], aX1);
    }
    __syncthreads();
    // state update (post-update x overwrites sX; outputs below used pre-update x)
    #pragma unroll
    for (int r = 0; r < 8; r++) {
      const int m = r + hi*8, n = w*16 + l15;
      sX[m*DXD + n]      = (bf16)(aX0[r] + bxv);
      sX[(m+16)*DXD + n] = (bf16)(aX1[r] + bxv);
    }
    if (w < 12) {
      #pragma unroll
      for (int r = 0; r < 8; r++) {
        const int m = r + hi*8;
        if (w < 8) {
          const int j = w*16 + l15;
          outy[((size_t)m*TT + t)*DYH + j]      = aY0[r] + byzv;
          outy[((size_t)(m+16)*TT + t)*DYH + j] = aY1[r] + byzv;
        } else {
          const int j = (w-8)*16 + l15;
          outz[((size_t)m*TT + t)*DZH + j]      = aY0[r] + byzv;
          outz[((size_t)(m+16)*TT + t)*DZH + j] = aY1[r] + byzv;
        }
      }
    }
    __syncthreads();
  }
}

// ---------------------------------------------------------------------------
extern "C" void kernel_launch(void* const* d_in, const int* in_sizes, int n_in,
                              void* d_out, int out_size, void* d_ws, size_t ws_size,
                              hipStream_t stream) {
  (void)in_sizes; (void)n_in; (void)out_size; (void)ws_size;
  const float* y    = (const float*)d_in[0];
  const float* u    = (const float*)d_in[1];
  const float* W_A  = (const float*)d_in[2];
  const float* b_A  = (const float*)d_in[3];
  const float* W_K  = (const float*)d_in[4];
  const float* b_K  = (const float*)d_in[5];
  const float* W_B  = (const float*)d_in[6];
  const float* b_B  = (const float*)d_in[7];
  const float* W_Cy = (const float*)d_in[8];
  const float* b_Cy = (const float*)d_in[9];
  const float* W_Cz = (const float*)d_in[10];
  const float* b_Cz = (const float*)d_in[11];

  char* wsb = (char*)d_ws;
  size_t off = 0;
  auto take = [&](size_t bytes) -> void* {
    void* p = wsb + off;
    off += (bytes + 255) & ~(size_t)255;
    return p;
  };
  bf16*  WXF    = (bf16*) take((size_t)DXD*448*2);     // 229 KB
  bf16*  WYZ    = (bf16*) take((size_t)192*DXD*2);     //  98 KB
  float* biasX  = (float*)take(DXD*4);
  float* biasYZ = (float*)take(192*4);
  float* P0     = (float*)take((size_t)DXD*DXD*4);     // 256 KB
  float* P1     = (float*)take((size_t)DXD*DXD*4);     // 256 KB
  bf16*  MPT    = (bf16*) take((size_t)DXD*DXD*2);     // 128 KB
  float* Dc     = (float*)take((size_t)NC*NB*DXD*4);   //   4 MB
  bf16*  XS     = (bf16*) take((size_t)NC*NB*DXD*2);   //   2 MB

  build_weights<<<448, 448, 0, stream>>>(W_A, b_A, W_K, b_K, W_B, b_B,
                                         W_Cy, b_Cy, W_Cz, b_Cz,
                                         WXF, WYZ, biasX, biasYZ, P0);
  // (W_A)^64 via 6 row-major squarings; result (even count) lands in P0.
  for (int i = 0; i < 6; i++) {
    const float* src = (i & 1) ? P1 : P0;
    float*       dst = (i & 1) ? P0 : P1;
    square256<<<DXD, DXD, 0, stream>>>(src, dst);
  }
  cvt256<<<DXD, 256, 0, stream>>>(P0, MPT);

  phase1<<<NC, 512, 0, stream>>>(y, u, WXF, biasX, Dc);
  combine<<<1, 512, 0, stream>>>(MPT, Dc, XS);
  phase3<<<NC, 512, 0, stream>>>(y, u, WXF, WYZ, biasX, biasYZ, XS, (float*)d_out);
}
